// UniModalLF_55465207661175
// MI455X (gfx1250) — matrix operand order repaired and checked by reading, not compiled, not run on hardware
//
#include <hip/hip_runtime.h>
#include <hip/hip_bf16.h>
#include <math.h>

// ---------------- problem constants ----------------
#define B_    128
#define T_    99
#define TM1_  98
#define DX_   1024
#define DZ_   64
#define DY_   32
#define DH_   256
#define DA_   9
#define HID_  512
#define MBT_  (B_*T_)          // 12672 rows for encoder/decoder GEMMs
#define KLSTM 352              // 73 (z|u) + 256 (h), padded to mult of 32
#define KTIN  128              // 105 (z|u|y), padded to mult of 32

#define LOG_X_STD (-2.302585093f)
#define LOG_Y_STD (-2.302585093f)

typedef __attribute__((ext_vector_type(16))) __bf16 v16bf;
typedef __attribute__((ext_vector_type(8)))  float  v8f;

// ---------------- deterministic hash RNG (splitmix64 + Box-Muller) --------
__device__ __forceinline__ float rng_normal(unsigned long long seed,
                                            unsigned long long idx) {
  unsigned long long z = seed + idx * 0x9E3779B97F4A7C15ULL;
  z = (z ^ (z >> 30)) * 0xBF58476D1CE4E5B9ULL;
  z = (z ^ (z >> 27)) * 0x94D049BB133111EBULL;
  z ^= (z >> 31);
  unsigned int u1 = (unsigned int)(z & 0xFFFFFFFFULL);
  unsigned int u2 = (unsigned int)(z >> 32);
  float f1 = ((float)u1 + 1.0f) * (1.0f / 4294967296.0f);
  float f2 = ((float)u2) * (1.0f / 4294967296.0f);
  float r = sqrtf(-2.0f * logf(f1));
  return r * cosf(6.28318530718f * f2);
}

// ---------------- WMMA bf16 GEMM: C = act(A[MxK] @ W[KxN] + bias) ---------
// Requirements (all call sites satisfy them): M %128==0, N %64==0, K %32==0.
// A: fp32 row-major. W: bf16 row-major (pre-converted). bias: fp32 length N.
// Workgroup tile 128x64, 8 waves; each wave owns a 2x2 grid of 16x16 tiles
// (4 accumulators) so every A/B fragment is reused twice: 8 ds_load_b128 per
// 4 v_wmma_f32_16x16x32_bf16.
// mode 0: C = v          (linear)
// mode 1: C = relu(v)
// mode 2: decoder final:  C[idx]   = v + 0.1*N(seed,idx)       (xs_hat)
//                         C2[2idx] = v, C2[2idx+1] = LOG_X_STD (x_hat_params)
#define TBM 128
#define TBN 64
#define TBK 32
#define LDA_S 40
#define LDB_S 72

__global__ __launch_bounds__(256) void wmma_gemm(
    const float* __restrict__ A, const __bf16* __restrict__ W,
    const float* __restrict__ bias, float* __restrict__ C,
    float* __restrict__ C2, int M, int N, int K, int mode,
    unsigned long long seed) {
  __shared__ __bf16 As[TBM * LDA_S];   // 128x32 (+pad)
  __shared__ __bf16 Bs[TBK * LDB_S];   // 32x64 (+pad)

  const int tid  = threadIdx.x;
  const int lane = tid & 31;
  const int wave = tid >> 5;                  // 8 waves (wave32)
  const int bm = blockIdx.y * TBM;
  const int bn = blockIdx.x * TBN;

  const int wm = (wave >> 1) * 32;            // 0/32/64/96: wave's 32-row strip
  const int wn = (wave & 1) * 32;             // 0/32:       wave's 32-col strip
  const int half = lane >> 4;                 // 0: lanes 0-15, 1: lanes 16-31
  const int l16  = lane & 15;

  v8f acc00 = {}, acc01 = {}, acc10 = {}, acc11 = {};

  for (int k0 = 0; k0 < K; k0 += TBK) {
    // A tile: 128x32 fp32 -> bf16 LDS (4096 elems / 256 thr = 16 each)
    for (int i = tid; i < TBM * TBK; i += 256) {
      int r = i >> 5, c = i & 31;
      As[r * LDA_S + c] = (__bf16)A[(size_t)(bm + r) * K + k0 + c];
      if (k0 + TBK < K)
        __builtin_prefetch(&A[(size_t)(bm + r) * K + k0 + TBK + c], 0, 0);
    }
    // B tile: 32x64 bf16 LDS
    for (int i = tid; i < TBK * TBN; i += 256) {
      int r = i >> 6, c = i & 63;
      Bs[r * LDB_S + c] = W[(size_t)(k0 + r) * N + bn + c];
    }
    __syncthreads();

    // B fragments: lane -> N (mod 16); element e -> K (same order as A; half selects +8)
    v16bf b0, b1;
    {
#pragma unroll
      for (int e = 0; e < 16; ++e) {
        int bk = ((e < 8) ? e : (16 + (e - 8))) + 8 * half;
        b0[e] = Bs[bk * LDB_S + wn + l16];
        b1[e] = Bs[bk * LDB_S + wn + 16 + l16];
      }
    }
    // A fragments: lane -> M (mod 16); VGPR j holds K pair; half selects +8
    v16bf a0, a1;
#pragma unroll
    for (int j = 0; j < 8; ++j) {
      int kk = ((j < 4) ? (2 * j) : (16 + 2 * (j - 4))) + 8 * half;
      a0[2 * j]     = As[(wm + l16) * LDA_S + kk];
      a0[2 * j + 1] = As[(wm + l16) * LDA_S + kk + 1];
      a1[2 * j]     = As[(wm + 16 + l16) * LDA_S + kk];
      a1[2 * j + 1] = As[(wm + 16 + l16) * LDA_S + kk + 1];
    }

    acc00 = __builtin_amdgcn_wmma_f32_16x16x32_bf16(
        false, a0, false, b0, (short)0, acc00, false, false);
    acc01 = __builtin_amdgcn_wmma_f32_16x16x32_bf16(
        false, a0, false, b1, (short)0, acc01, false, false);
    acc10 = __builtin_amdgcn_wmma_f32_16x16x32_bf16(
        false, a1, false, b0, (short)0, acc10, false, false);
    acc11 = __builtin_amdgcn_wmma_f32_16x16x32_bf16(
        false, a1, false, b1, (short)0, acc11, false, false);
    __syncthreads();
  }

  // Epilogue. C/D layout: VGPR r -> M = r + 8*half ; N = lane&15.
  const int col0 = bn + wn + l16;
  const int col1 = col0 + 16;
  const float bc0 = bias ? bias[col0] : 0.0f;
  const float bc1 = bias ? bias[col1] : 0.0f;
#pragma unroll
  for (int r = 0; r < 8; ++r) {
    int row0 = bm + wm + r + 8 * half;
    int row1 = row0 + 16;
    float v00 = acc00[r] + bc0;   // (row0, col0)
    float v01 = acc01[r] + bc1;   // (row0, col1)
    float v10 = acc10[r] + bc0;   // (row1, col0)
    float v11 = acc11[r] + bc1;   // (row1, col1)
    if (mode == 1) {
      v00 = fmaxf(v00, 0.0f); v01 = fmaxf(v01, 0.0f);
      v10 = fmaxf(v10, 0.0f); v11 = fmaxf(v11, 0.0f);
    }
    if (mode == 2) {
      size_t i00 = (size_t)row0 * N + col0, i01 = (size_t)row0 * N + col1;
      size_t i10 = (size_t)row1 * N + col0, i11 = (size_t)row1 * N + col1;
      C2[i00 * 2] = v00; C2[i00 * 2 + 1] = LOG_X_STD;
      C2[i01 * 2] = v01; C2[i01 * 2 + 1] = LOG_X_STD;
      C2[i10 * 2] = v10; C2[i10 * 2 + 1] = LOG_X_STD;
      C2[i11 * 2] = v11; C2[i11 * 2 + 1] = LOG_X_STD;
      C[i00] = v00 + 0.1f * rng_normal(seed, i00);
      C[i01] = v01 + 0.1f * rng_normal(seed, i01);
      C[i10] = v10 + 0.1f * rng_normal(seed, i10);
      C[i11] = v11 + 0.1f * rng_normal(seed, i11);
    } else {
      C[(size_t)row0 * N + col0] = v00;
      C[(size_t)row0 * N + col1] = v01;
      C[(size_t)row1 * N + col0] = v10;
      C[(size_t)row1 * N + col1] = v11;
    }
  }
}

// ---------------- small setup / pointwise kernels -------------------------
__global__ void cvt_bf16(__bf16* dst, const float* src, int n) {
  int i = blockIdx.x * blockDim.x + threadIdx.x;
  if (i < n) dst[i] = (__bf16)src[i];
}

__global__ void pack_fused_lstm(__bf16* dst, const float* Wih, const float* Whh) {
  int i = blockIdx.x * blockDim.x + threadIdx.x;
  if (i >= KLSTM * 1024) return;
  int r = i >> 10, c = i & 1023;
  float v = 0.0f;
  if (r < 73) v = Wih[r * 1024 + c];
  else if (r < 329) v = Whh[(r - 73) * 1024 + c];
  dst[i] = (__bf16)v;
}

__global__ void pack_tw1(__bf16* dst, const float* tW1) {
  int i = blockIdx.x * blockDim.x + threadIdx.x;
  if (i >= KTIN * HID_) return;
  int r = i / HID_, c = i % HID_;
  dst[i] = (__bf16)((r < 105) ? tW1[r * HID_ + c] : 0.0f);
}

__global__ void add_vec(float* dst, const float* a, const float* b, int n) {
  int i = blockIdx.x * blockDim.x + threadIdx.x;
  if (i < n) dst[i] = a[i] + b[i];
}

__global__ void zero_f32(float* dst, int n) {
  int i = blockIdx.x * blockDim.x + threadIdx.x;
  if (i < n) dst[i] = 0.0f;
}

// build in1 = [zs_t | u_t | h | 0pad]  (B x 352)
__global__ void pack_in1(float* in1, const float* zs_cur, const float* u,
                         const float* h, int t) {
  int i = blockIdx.x * blockDim.x + threadIdx.x;
  if (i >= B_ * KLSTM) return;
  int b = i / KLSTM, c = i % KLSTM;
  float v = 0.0f;
  if (c < 64)       v = zs_cur[b * DZ_ + c];
  else if (c < 73)  v = u[((size_t)b * T_ + t) * DA_ + (c - 64)];
  else if (c < 329) v = h[b * DH_ + (c - 73)];
  in1[i] = v;
}

__global__ void lstm_pw(const float* gates, float* h, float* c,
                        float* hs_out, float* cs_out) {
  int i = blockIdx.x * blockDim.x + threadIdx.x;
  if (i >= B_ * DH_) return;
  int b = i / DH_, j = i % DH_;
  const float* g = gates + (size_t)b * 1024;
  float ig = g[j], fg = g[256 + j], gg = g[512 + j], og = g[768 + j];
  float c_old = c[i], h_old = h[i];
  hs_out[i] = h_old;                     // carry BEFORE update
  cs_out[i] = c_old;
  float si = 1.0f / (1.0f + expf(-ig));
  float sf = 1.0f / (1.0f + expf(-fg));
  float so = 1.0f / (1.0f + expf(-og));
  float c1 = sf * c_old + si * tanhf(gg);
  float h1 = so * tanhf(c1);
  c[i] = c1;
  h[i] = h1;
}

__global__ void sample_y(const float* y_p, float* ys_b, float* yparams_out,
                         float* ys_out, int t, unsigned long long seed) {
  int i = blockIdx.x * blockDim.x + threadIdx.x;
  if (i >= B_ * DY_) return;
  int b = i / DY_, d = i % DY_;
  float mu = y_p[b * 64 + 2 * d];
  float ls = y_p[b * 64 + 2 * d + 1];
  float v = mu + expf(ls) * rng_normal(seed, (unsigned long long)t * B_ * DY_ + i);
  ys_b[i] = v;
  size_t o = ((size_t)t * B_ + b) * DY_ + d;
  yparams_out[2 * o] = mu;
  yparams_out[2 * o + 1] = ls;
  ys_out[o] = v;
}

// tin = [zs_t | u_{t+1} | ys | 0pad]  (B x 128)
__global__ void pack_tin(float* tin, const float* zs_cur, const float* u,
                         const float* ys_b, int t) {
  int i = blockIdx.x * blockDim.x + threadIdx.x;
  if (i >= B_ * KTIN) return;
  int b = i / KTIN, c = i % KTIN;
  float v = 0.0f;
  if (c < 64)       v = zs_cur[b * DZ_ + c];
  else if (c < 73)  v = u[((size_t)b * T_ + (t + 1)) * DA_ + (c - 64)];
  else if (c < 105) v = ys_b[b * DY_ + (c - 73)];
  tin[i] = v;
}

__global__ void fuse_z(const float* ztr, const float* zm, float* zs_cur,
                       float* zs_all, float* zparams_out, float* priorz_out,
                       float* zs_out, int t, unsigned long long seed) {
  int i = blockIdx.x * blockDim.x + threadIdx.x;
  if (i >= B_ * DZ_) return;
  int b = i / DZ_, d = i % DZ_;
  float mu1 = ztr[b * 128 + 2 * d];
  float ls1 = ztr[b * 128 + 2 * d + 1];
  float mu2 = zm[((size_t)b * T_ + (t + 1)) * 128 + 2 * d];
  float ls2 = zm[((size_t)b * T_ + (t + 1)) * 128 + 2 * d + 1];
  float p1 = expf(-2.0f * ls1), p2 = expf(-2.0f * ls2);
  float var = 1.0f / (p1 + p2);
  float mu = var * (mu1 * p1 + mu2 * p2);
  float lsf = 0.5f * logf(var);
  size_t o = ((size_t)(t + 1) * B_ + b) * DZ_ + d;
  zparams_out[2 * o] = mu;
  zparams_out[2 * o + 1] = lsf;
  if (t + 2 < T_) {
    size_t o2 = ((size_t)(t + 2) * B_ + b) * DZ_ + d;
    priorz_out[2 * o2] = mu;
    priorz_out[2 * o2 + 1] = lsf;
  }
  float z1 = mu + expf(lsf) * rng_normal(seed, (unsigned long long)t * B_ * DZ_ + i);
  zs_cur[i] = z1;
  zs_all[o] = z1;
  zs_out[o] = z1;
}

__global__ void init_t0(const float* zm, float* zs_cur, float* zs_all,
                        float* zparams_out, float* priorz_out, float* zs_out,
                        unsigned long long seed) {
  int i = blockIdx.x * blockDim.x + threadIdx.x;
  if (i >= B_ * DZ_) return;
  int b = i / DZ_, d = i % DZ_;
  float mu = zm[((size_t)b * T_) * 128 + 2 * d];
  float ls = zm[((size_t)b * T_) * 128 + 2 * d + 1];
  size_t o0 = (size_t)b * DZ_ + d;                  // t = 0
  zparams_out[2 * o0] = mu;  zparams_out[2 * o0 + 1] = ls;
  priorz_out[2 * o0] = 0.0f; priorz_out[2 * o0 + 1] = 0.0f;
  size_t o1 = ((size_t)B_ + b) * DZ_ + d;           // prior_z[1] = z_params[0]
  priorz_out[2 * o1] = mu;   priorz_out[2 * o1 + 1] = ls;
  float z0 = mu + expf(ls) * rng_normal(seed, (unsigned long long)i);
  zs_cur[i] = z0;
  zs_all[o0] = z0;
  zs_out[o0] = z0;
}

__global__ void prior_y_fill(const int* labels, const float* emb, float* out) {
  int i = blockIdx.x * blockDim.x + threadIdx.x;
  if (i >= TM1_ * B_ * DY_) return;
  int d = i % DY_;
  int b = (i / DY_) % B_;
  int obj = labels[b * 3 + 0] + labels[b * 3 + 1] + labels[b * 3 + 2];
  float mu = tanhf(emb[obj * DY_ + d]);
  out[2 * i] = mu;
  out[2 * i + 1] = LOG_Y_STD;
}

__global__ void xf_copy(const float* x, float* out) {
  int i = blockIdx.x * blockDim.x + threadIdx.x;
  if (i >= T_ * B_ * DX_) return;
  int d = i % DX_;
  int b = (i / DX_) % B_;
  int t = i / (B_ * DX_);
  out[i] = x[((size_t)b * T_ + t) * DX_ + d];
}

// ---------------- host orchestration --------------------------------------
extern "C" void kernel_launch(void* const* d_in, const int* in_sizes, int n_in,
                              void* d_out, int out_size, void* d_ws,
                              size_t ws_size, hipStream_t stream) {
  (void)in_sizes; (void)n_in; (void)out_size; (void)ws_size;
  const float* x      = (const float*)d_in[0];
  const float* u      = (const float*)d_in[1];
  const int*   labels = (const int*)d_in[2];
  const float* emb    = (const float*)d_in[4];
  const float* eW1 = (const float*)d_in[5];  const float* eb1 = (const float*)d_in[6];
  const float* eW2 = (const float*)d_in[7];  const float* eb2 = (const float*)d_in[8];
  const float* eW3 = (const float*)d_in[9];  const float* eb3 = (const float*)d_in[10];
  const float* dW1 = (const float*)d_in[11]; const float* db1 = (const float*)d_in[12];
  const float* dW2 = (const float*)d_in[13]; const float* db2 = (const float*)d_in[14];
  const float* dW3 = (const float*)d_in[15]; const float* db3 = (const float*)d_in[16];
  const float* Wih = (const float*)d_in[17]; const float* Whh = (const float*)d_in[18];
  const float* bih = (const float*)d_in[19]; const float* bhh = (const float*)d_in[20];
  const float* Wy  = (const float*)d_in[21]; const float* by  = (const float*)d_in[22];
  const float* tW1 = (const float*)d_in[23]; const float* tb1 = (const float*)d_in[24];
  const float* tW2 = (const float*)d_in[25]; const float* tb2 = (const float*)d_in[26];
  const float* tW3 = (const float*)d_in[27]; const float* tb3 = (const float*)d_in[28];

  float* out = (float*)d_out;
  // output layout (flat, return order)
  const size_t SZ_PRIOR_Y = (size_t)TM1_ * B_ * DY_ * 2;
  const size_t SZ_YS      = (size_t)TM1_ * B_ * DY_;
  const size_t SZ_PZ      = (size_t)T_ * B_ * DZ_ * 2;
  const size_t SZ_ZS      = (size_t)T_ * B_ * DZ_;
  const size_t SZ_XH      = (size_t)T_ * B_ * DX_;
  const size_t SZ_HS      = (size_t)TM1_ * B_ * DH_;
  size_t o = 0;
  float* o_prior_y = out + o; o += SZ_PRIOR_Y;
  float* o_yparams = out + o; o += SZ_PRIOR_Y;
  float* o_ys      = out + o; o += SZ_YS;
  float* o_prior_z = out + o; o += SZ_PZ;
  float* o_zparams = out + o; o += SZ_PZ;
  float* o_zs      = out + o; o += SZ_ZS;
  float* o_xs_hat  = out + o; o += SZ_XH;
  float* o_xhp     = out + o; o += 2 * SZ_XH;
  float* o_xf      = out + o; o += SZ_XH;
  float* o_hs      = out + o; o += SZ_HS;
  float* o_cs      = out + o; o += SZ_HS;

  // ---- workspace bump allocator ----
  char* ws = (char*)d_ws;
  size_t off = 0;
  auto alloc = [&](size_t bytes) -> void* {
    void* p = (void*)(ws + off);
    off += (bytes + 255) & ~(size_t)255;
    return p;
  };
  __bf16* eW1b = (__bf16*)alloc((size_t)1024 * 512 * 2);
  __bf16* eW2b = (__bf16*)alloc((size_t)512 * 512 * 2);
  __bf16* eW3b = (__bf16*)alloc((size_t)512 * 128 * 2);
  __bf16* dW1b = (__bf16*)alloc((size_t)64 * 512 * 2);
  __bf16* dW2b = (__bf16*)alloc((size_t)512 * 512 * 2);
  __bf16* dW3b = (__bf16*)alloc((size_t)512 * 1024 * 2);
  __bf16* Wfb  = (__bf16*)alloc((size_t)KLSTM * 1024 * 2);
  __bf16* Wyb  = (__bf16*)alloc((size_t)256 * 64 * 2);
  __bf16* tW1b = (__bf16*)alloc((size_t)KTIN * 512 * 2);
  __bf16* tW2b = (__bf16*)alloc((size_t)512 * 512 * 2);
  __bf16* tW3b = (__bf16*)alloc((size_t)512 * 128 * 2);
  float* bsum   = (float*)alloc(1024 * 4);
  float* buf1   = (float*)alloc((size_t)MBT_ * 512 * 4);
  float* buf2   = (float*)alloc((size_t)MBT_ * 512 * 4);
  float* zm     = (float*)alloc((size_t)MBT_ * 128 * 4);
  float* zs_all = (float*)alloc((size_t)MBT_ * 64 * 4);
  float* in1    = (float*)alloc((size_t)B_ * KLSTM * 4);
  float* gates  = (float*)alloc((size_t)B_ * 1024 * 4);
  float* hbuf   = (float*)alloc((size_t)B_ * DH_ * 4);
  float* cbuf   = (float*)alloc((size_t)B_ * DH_ * 4);
  float* y_p    = (float*)alloc((size_t)B_ * 64 * 4);
  float* ys_b   = (float*)alloc((size_t)B_ * DY_ * 4);
  float* tin    = (float*)alloc((size_t)B_ * KTIN * 4);
  float* t1b    = (float*)alloc((size_t)B_ * 512 * 4);
  float* t2b    = (float*)alloc((size_t)B_ * 512 * 4);
  float* ztrb   = (float*)alloc((size_t)B_ * 128 * 4);
  float* zs_cur = (float*)alloc((size_t)B_ * DZ_ * 4);

  auto grid1 = [](int n) { return dim3((unsigned)((n + 255) / 256)); };
  const dim3 blk(256);

  // ---- weight conversion / packing ----
  cvt_bf16<<<grid1(1024 * 512), blk, 0, stream>>>(eW1b, eW1, 1024 * 512);
  cvt_bf16<<<grid1(512 * 512), blk, 0, stream>>>(eW2b, eW2, 512 * 512);
  cvt_bf16<<<grid1(512 * 128), blk, 0, stream>>>(eW3b, eW3, 512 * 128);
  cvt_bf16<<<grid1(64 * 512),  blk, 0, stream>>>(dW1b, dW1, 64 * 512);
  cvt_bf16<<<grid1(512 * 512), blk, 0, stream>>>(dW2b, dW2, 512 * 512);
  cvt_bf16<<<grid1(512 * 1024), blk, 0, stream>>>(dW3b, dW3, 512 * 1024);
  cvt_bf16<<<grid1(256 * 64),  blk, 0, stream>>>(Wyb, Wy, 256 * 64);
  cvt_bf16<<<grid1(512 * 512), blk, 0, stream>>>(tW2b, tW2, 512 * 512);
  cvt_bf16<<<grid1(512 * 128), blk, 0, stream>>>(tW3b, tW3, 512 * 128);
  pack_fused_lstm<<<grid1(KLSTM * 1024), blk, 0, stream>>>(Wfb, Wih, Whh);
  pack_tw1<<<grid1(KTIN * HID_), blk, 0, stream>>>(tW1b, tW1);
  add_vec<<<grid1(1024), blk, 0, stream>>>(bsum, bih, bhh, 1024);
  zero_f32<<<grid1(B_ * DH_), blk, 0, stream>>>(hbuf, B_ * DH_);
  zero_f32<<<grid1(B_ * DH_), blk, 0, stream>>>(cbuf, B_ * DH_);

  // ---- encoder: x -> zm_params  (M = 12672 rows, (b,t) order) ----
  wmma_gemm<<<dim3(512 / TBN, MBT_ / TBM), blk, 0, stream>>>(
      x, eW1b, eb1, buf1, nullptr, MBT_, 512, 1024, 1, 0ULL);
  wmma_gemm<<<dim3(512 / TBN, MBT_ / TBM), blk, 0, stream>>>(
      buf1, eW2b, eb2, buf2, nullptr, MBT_, 512, 512, 1, 0ULL);
  wmma_gemm<<<dim3(128 / TBN, MBT_ / TBM), blk, 0, stream>>>(
      buf2, eW3b, eb3, zm, nullptr, MBT_, 128, 512, 0, 0ULL);

  // ---- t = 0 init + static priors ----
  init_t0<<<grid1(B_ * DZ_), blk, 0, stream>>>(zm, zs_cur, zs_all, o_zparams,
                                               o_prior_z, o_zs, 0x5151ULL);
  prior_y_fill<<<grid1(TM1_ * B_ * DY_), blk, 0, stream>>>(labels, emb, o_prior_y);
  xf_copy<<<grid1(T_ * B_ * DX_), blk, 0, stream>>>(x, o_xf);

  // ---- sequential scan over 98 steps (M = 128 -> grid.y = 1) ----
  for (int t = 0; t < TM1_; ++t) {
    pack_in1<<<grid1(B_ * KLSTM), blk, 0, stream>>>(in1, zs_cur, u, hbuf, t);
    wmma_gemm<<<dim3(1024 / TBN, B_ / TBM), blk, 0, stream>>>(
        in1, Wfb, bsum, gates, nullptr, B_, 1024, KLSTM, 0, 0ULL);
    lstm_pw<<<grid1(B_ * DH_), blk, 0, stream>>>(
        gates, hbuf, cbuf, o_hs + (size_t)t * B_ * DH_,
        o_cs + (size_t)t * B_ * DH_);
    wmma_gemm<<<dim3(64 / TBN, B_ / TBM), blk, 0, stream>>>(
        hbuf, Wyb, by, y_p, nullptr, B_, 64, 256, 0, 0ULL);
    sample_y<<<grid1(B_ * DY_), blk, 0, stream>>>(y_p, ys_b, o_yparams, o_ys, t,
                                                  0xABCDULL);
    pack_tin<<<grid1(B_ * KTIN), blk, 0, stream>>>(tin, zs_cur, u, ys_b, t);
    wmma_gemm<<<dim3(512 / TBN, B_ / TBM), blk, 0, stream>>>(
        tin, tW1b, tb1, t1b, nullptr, B_, 512, KTIN, 1, 0ULL);
    wmma_gemm<<<dim3(512 / TBN, B_ / TBM), blk, 0, stream>>>(
        t1b, tW2b, tb2, t2b, nullptr, B_, 512, 512, 1, 0ULL);
    wmma_gemm<<<dim3(128 / TBN, B_ / TBM), blk, 0, stream>>>(
        t2b, tW3b, tb3, ztrb, nullptr, B_, 128, 512, 0, 0ULL);
    fuse_z<<<grid1(B_ * DZ_), blk, 0, stream>>>(ztrb, zm, zs_cur, zs_all,
                                                o_zparams, o_prior_z, o_zs, t,
                                                0x7777ULL);
  }

  // ---- decoder: zs (T,B) order -> xs_hat / x_hat_params ----
  wmma_gemm<<<dim3(512 / TBN, MBT_ / TBM), blk, 0, stream>>>(
      zs_all, dW1b, db1, buf1, nullptr, MBT_, 512, 64, 1, 0ULL);
  wmma_gemm<<<dim3(512 / TBN, MBT_ / TBM), blk, 0, stream>>>(
      buf1, dW2b, db2, buf2, nullptr, MBT_, 512, 512, 1, 0ULL);
  wmma_gemm<<<dim3(1024 / TBN, MBT_ / TBM), blk, 0, stream>>>(
      buf2, dW3b, db3, o_xs_hat, o_xhp, MBT_, 1024, 512, 2, 0x9999ULL);
}
